// AttentionBlock_17042430230929
// MI455X (gfx1250) — matrix operand order repaired and checked by reading, not compiled, not run on hardware
//
#include <hip/hip_runtime.h>
#include <math.h>

typedef __attribute__((ext_vector_type(16))) _Float16 v16h;
typedef __attribute__((ext_vector_type(8)))  float    v8f;

#define NROW 8192
#define DDIM 256

union FragH { v16h v; _Float16 h[16]; unsigned u[8]; };

__device__ __forceinline__ v8f vzero8() {
  v8f z = {0.f,0.f,0.f,0.f,0.f,0.f,0.f,0.f};
  return z;
}

// A fragment (16x32 f16, MxK): row m0+(lane&15); K pairs per ISA layout:
//  lanes 0-15: v0..v3 -> K 0..7,  v4..v7 -> K 16..23
//  lanes16-31: v0..v3 -> K 8..15, v4..v7 -> K 24..31
__device__ __forceinline__ v16h load_frag_A(const _Float16* base, int ld,
                                            int m0, int k0, int lane) {
  const int g = (lane >> 4) & 1;
  const int r = lane & 15;
  FragH f;
  const _Float16* rp = base + (size_t)(m0 + r) * ld;
#pragma unroll
  for (int p = 0; p < 8; ++p) {
    int k = k0 + ((p >> 2) << 4) + (g << 3) + ((p & 3) << 1);
    f.u[p] = *(const unsigned*)(rp + k);
  }
  return f.v;
}

// B fragment (32x16 f16, KxN) from a transposed view: B[k][n] = M[(n0+n)*ld + k0+k]
//  lanes 0-15: col n=lane, K 0..7,16..23 ; lanes 16-31: K 8..15,24..31
__device__ __forceinline__ v16h load_frag_BT(const _Float16* base, int ld,
                                             int n0, int k0, int lane) {
  const int g = (lane >> 4) & 1;
  const int n = lane & 15;
  FragH f;
  const _Float16* rp = base + (size_t)(n0 + n) * ld + k0;
#pragma unroll
  for (int p = 0; p < 8; ++p)
    f.u[p] = *(const unsigned*)(rp + ((p >> 2) << 4) + (g << 3) + ((p & 3) << 1));
  return f.v;
}

__device__ __forceinline__ v8f wmma_f16(v16h a, v16h b, v8f c) {
  return __builtin_amdgcn_wmma_f32_16x16x32_f16(false, a, false, b,
                                                (short)0, c, false, false);
}

// ---------------- Kernel 0: f32 -> f16 convert ----------------
__global__ void cvt_f32_f16(const float* __restrict__ in,
                            _Float16* __restrict__ out, int n) {
  int i = (blockIdx.x * blockDim.x + threadIdx.x) * 4;
  if (i + 3 < n) {
    float4 v = *(const float4*)(in + i);
    union { unsigned long long u; _Float16 h[4]; } o;
    o.h[0] = (_Float16)v.x; o.h[1] = (_Float16)v.y;
    o.h[2] = (_Float16)v.z; o.h[3] = (_Float16)v.w;
    *(unsigned long long*)(out + i) = o.u;
  }
}

// ---------------- Kernel 1: q,k,v projections + L2 norm ----------------
// Block: 16 rows x full 256 cols. Wave w owns d-tiles {2w, 2w+1} of q,k,v.
// v is written TRANSPOSED (vT[d][n]) so the attention kernel can async-DMA
// contiguous 16B segments of it straight into LDS.
__global__ __launch_bounds__(256)
void qkv_kernel(const _Float16* __restrict__ x16,
                const _Float16* __restrict__ wq, const _Float16* __restrict__ wk,
                const _Float16* __restrict__ wv,
                const float* __restrict__ bq, const float* __restrict__ bk,
                const float* __restrict__ bv,
                _Float16* __restrict__ qn, _Float16* __restrict__ kn,
                _Float16* __restrict__ vT) {
  __shared__ float sQ[16];
  __shared__ float sK[16];
  __shared__ _Float16 sV[256 * 16];   // [d][row_local] staging for vT writeout
  const int tid = threadIdx.x, wave = tid >> 5, lane = tid & 31;
  const int g = (lane >> 4) & 1, ln = lane & 15;
  const int row0 = blockIdx.x * 16;

  v8f qa[2] = {vzero8(), vzero8()};
  v8f ka[2] = {vzero8(), vzero8()};
  v8f va[2] = {vzero8(), vzero8()};

  for (int ks = 0; ks < 8; ++ks) {
    const int k0 = ks * 32;
    const int d0 = wave * 32;
    v16h aX = load_frag_A(x16, DDIM, row0, k0, lane);   // shared by all 6 tiles
    v16h bQ0 = load_frag_BT(wq, DDIM, d0,      k0, lane);
    v16h bK0 = load_frag_BT(wk, DDIM, d0,      k0, lane);
    v16h bV0 = load_frag_BT(wv, DDIM, d0,      k0, lane);
    v16h bQ1 = load_frag_BT(wq, DDIM, d0 + 16, k0, lane);
    v16h bK1 = load_frag_BT(wk, DDIM, d0 + 16, k0, lane);
    v16h bV1 = load_frag_BT(wv, DDIM, d0 + 16, k0, lane);
    qa[0] = wmma_f16(aX, bQ0, qa[0]);
    ka[0] = wmma_f16(aX, bK0, ka[0]);
    va[0] = wmma_f16(aX, bV0, va[0]);
    qa[1] = wmma_f16(aX, bQ1, qa[1]);
    ka[1] = wmma_f16(aX, bK1, ka[1]);
    va[1] = wmma_f16(aX, bV1, va[1]);
  }

  if (tid < 16) { sQ[tid] = 0.f; sK[tid] = 0.f; }
  __syncthreads();

  float ssq[8], ssk[8];
#pragma unroll
  for (int i = 0; i < 8; ++i) { ssq[i] = 0.f; ssk[i] = 0.f; }
#pragma unroll
  for (int t = 0; t < 2; ++t) {
    const int d = (wave * 2 + t) * 16 + ln;
    const float bqv = bq[d], bkv = bk[d], bvv = bv[d];
#pragma unroll
    for (int i = 0; i < 8; ++i) {
      qa[t][i] += bqv; ka[t][i] += bkv; va[t][i] += bvv;
      ssq[i] += qa[t][i] * qa[t][i];
      ssk[i] += ka[t][i] * ka[t][i];
    }
  }
#pragma unroll
  for (int i = 0; i < 8; ++i) {
    atomicAdd(&sQ[g * 8 + i], ssq[i]);
    atomicAdd(&sK[g * 8 + i], ssk[i]);
  }
  __syncthreads();

#pragma unroll
  for (int i = 0; i < 8; ++i) {
    const int rl = g * 8 + i;
    const size_t row = (size_t)(row0 + rl);
    const float qi = 1.f / fmaxf(sqrtf(sQ[rl]), 1e-12f);
    const float ki = 1.f / fmaxf(sqrtf(sK[rl]), 1e-12f);
#pragma unroll
    for (int t = 0; t < 2; ++t) {
      const int d = (wave * 2 + t) * 16 + ln;
      qn[row * DDIM + d] = (_Float16)(qa[t][i] * qi);
      kn[row * DDIM + d] = (_Float16)(ka[t][i] * ki);
      sV[d * 16 + rl]    = (_Float16)(va[t][i]);
    }
  }
  __syncthreads();
  // coalesced transposed writeout: thread tid owns row d=tid of vT (16 halves)
  {
    const uint4* src = (const uint4*)&sV[tid * 16];
    uint4* dst = (uint4*)(vT + (size_t)tid * NROW + row0);
    dst[0] = src[0];
    dst[1] = src[1];
  }
}

// ---------------- Kernel 2: fused relu-similarity attention ----------------
// Block: 32 rows, full 256 output cols; sweep 8192 cols in chunks of 64.
// Wave w: S tile (rt = w&1, ct = w>>1); O tiles (rt = w&1, d-tiles (w>>1)*4..+3).
// v chunk is staged into LDS with GLOBAL_LOAD_ASYNC_TO_LDS_B128 (ASYNCcnt),
// issued one chunk ahead so the DMA overlaps the S-phase WMMAs.

__device__ __forceinline__ void stage_v_async(const _Float16* __restrict__ vT,
                                              _Float16* sVT, int col0, int tid) {
#pragma unroll
  for (int it = 0; it < 8; ++it) {
    const int seg = it * 256 + tid;       // 0..2047, 16B each
    const int d   = seg >> 3;             // 0..255
    const int k0  = (seg & 7) << 3;       // 0,8,..,56
    const unsigned long long gaddr =
        (unsigned long long)(size_t)(vT + (size_t)d * NROW + col0 + k0);
    const unsigned ldsaddr = (unsigned)(size_t)(sVT + d * 64 + k0);
    asm volatile("global_load_async_to_lds_b128 %0, %1, off"
                 :: "v"(ldsaddr), "v"(gaddr) : "memory");
  }
}

__global__ __launch_bounds__(256)
void attn_kernel(const _Float16* __restrict__ qn, const _Float16* __restrict__ kn,
                 const _Float16* __restrict__ vT, const float* __restrict__ x,
                 float* __restrict__ out) {
  __shared__ _Float16 sS[32 * 64];    // relu'd S chunk, [row][col], ld=64
  __shared__ _Float16 sVT[256 * 64];  // v chunk transposed, [d][col], ld=64
  __shared__ float sRS[32];           // row sums

  const int tid = threadIdx.x, wave = tid >> 5, lane = tid & 31;
  const int g = (lane >> 4) & 1, ln = lane & 15;
  const int row0 = blockIdx.x * 32;
  const int rt = wave & 1;          // row tile (both S and O phases)
  const int ct = wave >> 1;         // S col tile 0..3
  const int od0 = (wave >> 1) * 4;  // first O d-tile

  // async-stage chunk 0 while we preload q fragments
  stage_v_async(vT, sVT, 0, tid);

  v16h aQ[8];
#pragma unroll
  for (int ks = 0; ks < 8; ++ks)
    aQ[ks] = load_frag_A(qn, DDIM, row0 + rt * 16, ks * 32, lane);

  if (tid < 32) sRS[tid] = 0.f;

  v8f oacc[4] = {vzero8(), vzero8(), vzero8(), vzero8()};
  float rsv[8];
#pragma unroll
  for (int i = 0; i < 8; ++i) rsv[i] = 0.f;

  for (int c = 0; c < NROW / 64; ++c) {
    const int col0 = c * 64;

    // S tile = qn[rows] @ kn[cols]^T (K=256); preload all B frags -> one clause
    v16h bK[8];
#pragma unroll
    for (int ks = 0; ks < 8; ++ks)
      bK[ks] = load_frag_BT(kn, DDIM, col0 + ct * 16, ks * 32, lane);
    v8f sacc = vzero8();
#pragma unroll
    for (int ks = 0; ks < 8; ++ks)
      sacc = wmma_f16(aQ[ks], bK[ks], sacc);

    // relu + zero-diagonal + rowsum, spill to LDS as f16
#pragma unroll
    for (int i = 0; i < 8; ++i) {
      const int rl = rt * 16 + g * 8 + i;
      const int rowg = row0 + rl;
      const int colg = col0 + ct * 16 + ln;
      float s = sacc[i];
      s = (s > 0.f && rowg != colg) ? s : 0.f;
      rsv[i] += s;
      sS[rl * 64 + ct * 16 + ln] = (_Float16)s;
    }

    // my async DMA for this chunk must have landed; barrier covers all waves
    asm volatile("s_wait_asynccnt 0x0" ::: "memory");
    __syncthreads();

    // O += S @ v  (K = 64 over this chunk)
#pragma unroll
    for (int kk = 0; kk < 2; ++kk) {
      v16h a2 = load_frag_A(sS, 64, rt * 16, kk * 32, lane);
#pragma unroll
      for (int j = 0; j < 4; ++j) {
        v16h b2 = load_frag_BT(sVT, 64, (od0 + j) * 16, kk * 32, lane);
        oacc[j] = wmma_f16(a2, b2, oacc[j]);
      }
    }
    __syncthreads();  // O-phase reads done -> safe to overwrite sVT/sS

    if (c + 1 < NROW / 64) {
      stage_v_async(vT, sVT, col0 + 64, tid);          // overlaps next S phase
      // hint next chunk's kn rows toward the caches
      const int pr = tid >> 2, po = (tid & 3) * 64;
      __builtin_prefetch(kn + (size_t)(col0 + 64 + pr) * DDIM + po, 0, 3);
    }
  }

#pragma unroll
  for (int i = 0; i < 8; ++i)
    atomicAdd(&sRS[rt * 16 + g * 8 + i], rsv[i]);
  __syncthreads();

  // epilogue: out = O / max(rowsum, eps) + x
#pragma unroll
  for (int j = 0; j < 4; ++j) {
    const int d = (od0 + j) * 16 + ln;
#pragma unroll
    for (int i = 0; i < 8; ++i) {
      const int rl = rt * 16 + g * 8 + i;
      const size_t idx = (size_t)(row0 + rl) * DDIM + d;
      const float denom = fmaxf(sRS[rl], 1e-12f);
      out[idx] = oacc[j][i] / denom + x[idx];
    }
  }
}

extern "C" void kernel_launch(void* const* d_in, const int* in_sizes, int n_in,
                              void* d_out, int out_size, void* d_ws, size_t ws_size,
                              hipStream_t stream) {
  (void)in_sizes; (void)n_in; (void)out_size; (void)ws_size;
  const float* x  = (const float*)d_in[0];
  const float* Wq = (const float*)d_in[1];
  const float* bq = (const float*)d_in[2];
  const float* Wk = (const float*)d_in[3];
  const float* bk = (const float*)d_in[4];
  const float* Wv = (const float*)d_in[5];
  const float* bv = (const float*)d_in[6];
  float* out = (float*)d_out;

  char* ws = (char*)d_ws;
  size_t off = 0;
  _Float16* x16 = (_Float16*)(ws + off); off += (size_t)NROW * DDIM * 2;
  _Float16* w16q = (_Float16*)(ws + off); off += (size_t)DDIM * DDIM * 2;
  _Float16* w16k = (_Float16*)(ws + off); off += (size_t)DDIM * DDIM * 2;
  _Float16* w16v = (_Float16*)(ws + off); off += (size_t)DDIM * DDIM * 2;
  _Float16* qn  = (_Float16*)(ws + off); off += (size_t)NROW * DDIM * 2;
  _Float16* kn  = (_Float16*)(ws + off); off += (size_t)NROW * DDIM * 2;
  _Float16* vT  = (_Float16*)(ws + off); off += (size_t)NROW * DDIM * 2;

  const int NT = 256;
  cvt_f32_f16<<<(NROW * DDIM / 4 + NT - 1) / NT, NT, 0, stream>>>(x, x16, NROW * DDIM);
  cvt_f32_f16<<<(DDIM * DDIM / 4 + NT - 1) / NT, NT, 0, stream>>>(Wq, w16q, DDIM * DDIM);
  cvt_f32_f16<<<(DDIM * DDIM / 4 + NT - 1) / NT, NT, 0, stream>>>(Wk, w16k, DDIM * DDIM);
  cvt_f32_f16<<<(DDIM * DDIM / 4 + NT - 1) / NT, NT, 0, stream>>>(Wv, w16v, DDIM * DDIM);

  qkv_kernel<<<NROW / 16, 256, 0, stream>>>(x16, w16q, w16k, w16v,
                                            bq, bk, bv, qn, kn, vT);
  attn_kernel<<<NROW / 32, 256, 0, stream>>>(qn, kn, vT, x, out);
}
